// Student_Hyp_REC_34626026340596
// MI455X (gfx1250) — compile-verified, matrix operand order checked by reading, not run on hardware
//
#include <hip/hip_runtime.h>
#include <math.h>
#include <stdint.h>

#define DD  200
#define DP  208
#define LL  50
#define BB  4096
#define AA  14
#define SS  30

typedef __attribute__((ext_vector_type(2)))  float    v2f;
typedef __attribute__((ext_vector_type(8)))  float    v8f;
typedef __attribute__((ext_vector_type(16))) _Float16 v16h;

__device__ __forceinline__ float wave_reduce_sum(float v) {
#pragma unroll
  for (int m = 16; m > 0; m >>= 1) v += __shfl_xor(v, m, 32);
  return v;
}

// 256-thread (8-wave) block reduction; broadcasts result to all threads.
__device__ __forceinline__ float block_reduce_sum(float v, float* red, int tid) {
  v = wave_reduce_sum(v);
  if ((tid & 31) == 0) red[tid >> 5] = v;
  __syncthreads();
  float r;
  if (tid == 0) {
    r = red[0];
#pragma unroll
    for (int w = 1; w < 8; ++w) r += red[w];
    red[0] = r;
  }
  __syncthreads();
  r = red[0];
  __syncthreads();  // red[] reusable after return
  return r;
}

// ---------------- Kernel 0: anchors (tiny) ----------------
__global__ void k_anchors(const float* __restrict__ a_emb, const float* __restrict__ seed_w,
                          float* __restrict__ hyp, float* __restrict__ kle,
                          float* __restrict__ q2v, float* __restrict__ lfv) {
  __shared__ float sw[SS];
  __shared__ float red[8];
  const int a = blockIdx.x;
  const int tid = threadIdx.x;
  if (tid < SS) sw[tid] = seed_w[a * SS + tid];
  __syncthreads();
  float w = 0.f;
  if (tid < DD) {
    const float* base = a_emb + (size_t)a * SS * DD + tid;
#pragma unroll 5
    for (int s = 0; s < SS; ++s) w += base[s * DD] * sw[s];
  }
  float un2 = block_reduce_sum((tid < DD) ? w * w : 0.f, red, tid);
  // expmap0 + proj
  float un    = sqrtf(un2);
  float unc   = fmaxf(un, 1e-15f);
  float scale = tanhf(unc) / unc;
  float q     = scale * w;
  float pnraw = scale * un;
  float pnc   = fmaxf(pnraw, 1e-15f);
  const float maxn = 1.f - 1e-5f;
  float fac = (pnc > maxn) ? (maxn / pnc) : 1.f;
  q *= fac;
  float qn = (pnc > maxn) ? maxn : pnraw;
  float q2 = qn * qn;
  float kk = 2.f * q / (1.f + q2);
  if (tid < DP) {
    hyp[a * DP + tid] = (tid < DD) ? q : 0.f;
    kle[a * DP + tid] = (tid < DD) ? kk : 0.f;
  }
  float kn2 = block_reduce_sum((tid < DD) ? kk * kk : 0.f, red, tid);
  if (tid == 0) {
    q2v[a] = q2;
    lfv[a] = rsqrtf(fmaxf(1.f - kn2, 1e-7f));
  }
}

// ---------------- Kernel 1: x_avg (gather pass 1) ----------------
__global__ void k_xavg(const int* __restrict__ inp, const float* __restrict__ emb,
                       float* __restrict__ xavg) {
  __shared__ int idx[LL];
  const int b = blockIdx.x, tid = threadIdx.x;
  if (tid < LL) idx[tid] = inp[b * LL + tid];
  __syncthreads();
  if (tid < DP) {
    float s = 0.f;
    if (tid < DD) {
#pragma unroll 5
      for (int l = 0; l < LL; ++l) s += emb[(size_t)idx[l] * DD + tid];
      s *= (1.f / LL);
    }
    xavg[(size_t)b * DP + tid] = s;  // zero-padded cols 200..207
  }
}

// ---------------- Kernel 1b: zero-pad M into [DP x DP] (removes GEMM divergence) ----
__global__ void k_padM(const float* __restrict__ Mw, float* __restrict__ Mp) {
  int i = blockIdx.x * 256 + threadIdx.x;
  if (i < DP * DP) {
    int c = i / DP, k = i - c * DP;
    Mp[i] = (c < DD && k < DD) ? Mw[c * DD + k] : 0.f;
  }
}

// ---------------- Kernel 2: V = Xavg @ M^T via fp32 WMMA ----------------
// V[b,n] = sum_k Xavg[b,k] * Mp[n,k]. One 16x16 output tile per wave.
__global__ void k_vgemm(const float* __restrict__ xavg, const float* __restrict__ Mp,
                        float* __restrict__ Vm) {
  const int NT_N = DP / 16;                       // 13 tiles of N (cols 200..207 are junk pad)
  const int wid  = threadIdx.x >> 5;
  const int lane = threadIdx.x & 31;
  const int tile = blockIdx.x * 8 + wid;
  if (tile >= (BB / 16) * NT_N) return;           // uniform per wave
  const int tm = tile / NT_N, tn = tile % NT_N;
  const int m0 = tm * 16, n0 = tn * 16;
  const int h  = lane >> 4, lr = lane & 15;
  const int row = m0 + lr;                        // A-matrix row (always < BB)
  const int col = n0 + lr;                        // B-matrix column (always < DP)
  v8f acc = {};
#if __has_builtin(__builtin_amdgcn_wmma_f32_16x16x4_f32)
  // fp32 WMMA: A 16x4 (2 VGPRs: lanes 0-15 hold K={0,1}, lanes 16-31 K={2,3}); B mirrors.
  const float* arow = xavg + (size_t)row * DP + 2 * h;
  const float* brow = Mp + (size_t)col * DP + 2 * h;
  for (int k0 = 0; k0 < DD; k0 += 4) {
    v2f af = *(const v2f*)(arow + k0);   // 8B-aligned b64 load
    v2f bf = *(const v2f*)(brow + k0);   // 8B-aligned b64 load, no divergence
    acc = __builtin_amdgcn_wmma_f32_16x16x4_f32(false, af, false, bf, (short)0, acc,
                                                false, false);
  }
#else
  // f16 fallback (codegen-confirmed builtin), K padded with zeros via guards
  for (int k0 = 0; k0 < 224; k0 += 32) {
    v16h af = {}, bf = {};
#pragma unroll
    for (int v = 0; v < 8; ++v) {
      int kb = (v < 4) ? (2 * v + 8 * h) : (2 * v + 8 + 8 * h);
      int k  = k0 + kb;
      float a0 = (k < DP) ? xavg[(size_t)row * DP + k] : 0.f;
      float a1 = (k + 1 < DP) ? xavg[(size_t)row * DP + k + 1] : 0.f;
      af[2 * v]     = (_Float16)a0;
      af[2 * v + 1] = (_Float16)a1;
      float b0 = (k < DP) ? Mp[(size_t)col * DP + k] : 0.f;
      float b1 = (k + 1 < DP) ? Mp[(size_t)col * DP + k + 1] : 0.f;
      bf[2 * v]     = (_Float16)b0;
      bf[2 * v + 1] = (_Float16)b1;
    }
    acc = __builtin_amdgcn_wmma_f32_16x16x32_f16(false, af, false, bf, (short)0, acc,
                                                 false, false);
  }
#endif
#pragma unroll
  for (int r = 0; r < 8; ++r)
    Vm[(size_t)(m0 + r + 8 * h) * DP + n0 + lr] = acc[r];  // C/D layout: lanes16-31 = M+8
}

// ---------------- Kernel 3: fused attention + hyperbolic chain ----------------
__global__ void k_final(const int* __restrict__ inp, const float* __restrict__ emb,
                        const float* __restrict__ Vm,
                        const float* __restrict__ hyp, const float* __restrict__ kle,
                        const float* __restrict__ q2v, const float* __restrict__ lfv,
                        float* __restrict__ out_eu, float* __restrict__ out_pe,
                        float* __restrict__ out_ap) {
  __shared__ float xw[LL * DD];   // 40 KB: gathered word vectors
  __shared__ float vv[DP];        // v = M @ x_avg, later reused for p
  __shared__ float sc[64];        // scores -> softmax weights
  __shared__ float sdot[AA];
  __shared__ float sq2[AA], slf[AA];
  __shared__ float red[8];
  __shared__ int   idx[LL];
  const int b = blockIdx.x, tid = threadIdx.x;
  const int wid = tid >> 5, lane = tid & 31;

  if (tid < LL) idx[tid] = inp[b * LL + tid];
  if (tid < AA) { sq2[tid] = q2v[tid]; slf[tid] = lfv[tid]; }
  if (tid < DD) vv[tid] = Vm[(size_t)b * DP + tid];
  __syncthreads();

  // Async gather into LDS: GLOBAL_LOAD_ASYNC_TO_LDS_B128 (ASYNCcnt-tracked DMA path).
  // Each 16B chunk stays within one 800B embedding row (200 % 4 == 0) and is
  // 16B-aligned on both the global and LDS side.
  {
    const uint32_t lbase = (uint32_t)(uintptr_t)xw;  // flat LDS addr low bits = LDS offset
    for (int c = tid; c < (LL * DD) / 4; c += 256) {
      int l = (c * 4) / DD;
      int d = (c * 4) - l * DD;
      uint64_t ga = (uint64_t)(uintptr_t)(emb + (size_t)idx[l] * DD + d);
      uint32_t la = lbase + (uint32_t)(c * 16);
      asm volatile("global_load_async_to_lds_b128 %0, %1, off"
                   :: "v"(la), "v"(ga) : "memory");
    }
    asm volatile("s_wait_asynccnt 0x0" ::: "memory");
  }
  __syncthreads();

  // scores[l] = x_wrd[l,:] . v   (one wave per l-group; conflict-free LDS reads)
  for (int l = wid; l < LL; l += 8) {
    float s = 0.f;
    for (int d = lane; d < DD; d += 32) s += xw[l * DD + d] * vv[d];
    s = wave_reduce_sum(s);
    if (lane == 0) sc[l] = s;
  }
  __syncthreads();

  // softmax over L=50 within wave 0
  if (tid < 32) {
    float v0 = (tid < LL) ? sc[tid] : -3.4e38f;
    float v1 = (tid + 32 < LL) ? sc[tid + 32] : -3.4e38f;
    float m = fmaxf(v0, v1);
#pragma unroll
    for (int s = 16; s > 0; s >>= 1) m = fmaxf(m, __shfl_xor(m, s, 32));
    float e0 = (tid < LL) ? expf(v0 - m) : 0.f;
    float e1 = (tid + 32 < LL) ? expf(v1 - m) : 0.f;
    float sum = e0 + e1;
#pragma unroll
    for (int s = 16; s > 0; s >>= 1) sum += __shfl_xor(sum, s, 32);
    float inv = 1.f / sum;
    if (tid < LL) sc[tid] = e0 * inv;
    if (tid + 32 < LL) sc[tid + 32] = e1 * inv;
  }
  __syncthreads();

  // enc[d] = sum_l a[l] * x_wrd[l,d]
  float e = 0.f;
  if (tid < DD) {
#pragma unroll 5
    for (int l = 0; l < LL; ++l) e += sc[l] * xw[l * DD + tid];
  }
  float un2 = block_reduce_sum((tid < DD) ? e * e : 0.f, red, tid);

  // expmap0 + proj -> p (point on Poincare ball), p2 = ||p||^2
  float un    = sqrtf(un2);
  float unc   = fmaxf(un, 1e-15f);
  float scale = tanhf(unc) / unc;
  float p     = scale * e;
  float pnraw = scale * un;
  float pnc   = fmaxf(pnraw, 1e-15f);
  const float maxn = 1.f - 1e-5f;
  float fac = (pnc > maxn) ? (maxn / pnc) : 1.f;
  p *= fac;
  float pfn = (pnc > maxn) ? maxn : pnraw;
  float p2  = pfn * pfn;

  // anchor dots: one anchor per wave (2 barriers total)
  if (tid < DD) vv[tid] = p;  // reuse vv to hold p
  __syncthreads();
  for (int a = wid; a < AA; a += 8) {
    float s = 0.f;
    for (int d = lane; d < DD; d += 32) s += vv[d] * hyp[a * DP + d];
    s = wave_reduce_sum(s);
    if (lane == 0) sdot[a] = s;
  }
  __syncthreads();

  // per-anchor scalar chain (all threads redundantly; sqdist via 3 dot products)
  float pe[AA], ap[AA];
  float sum_pe = 0.f, sum_ap = 0.f;
#pragma unroll
  for (int a = 0; a < AA; ++a) {
    float xy    = -sdot[a];          // (-p) . q
    float y2    = sq2[a];
    float alpha = 1.f + 2.f * xy + y2;
    float beta  = 1.f - p2;
    float den   = fmaxf(1.f + 2.f * xy + p2 * y2, 1e-15f);
    float num2  = fmaxf(alpha * alpha * p2 + 2.f * alpha * beta * xy + beta * beta * y2, 0.f);
    float man   = sqrtf(num2) / den;
    float mc    = fminf(fmaxf(man, -1.f + 1e-7f), 1.f - 1e-7f);
    float ddist = logf((1.f + mc) / (1.f - mc));  // 2 * artanh
    float sqd   = ddist * ddist;
    float pex   = expf(-sqd - 0.05f);
    ap[a] = pex;  sum_ap += pex;
    float pr = slf[a] * pex;
    pe[a] = pr;   sum_pe += pr;
  }
  float ipe = 1.f / sum_pe, iap = 1.f / sum_ap;
#pragma unroll
  for (int a = 0; a < AA; ++a) { pe[a] *= ipe; ap[a] *= iap; }
  if (tid == 0) {
#pragma unroll
    for (int a = 0; a < AA; ++a) {
      out_pe[b * AA + a] = pe[a];
      out_ap[b * AA + a] = ap[a];
    }
  }

  // Klein barycenter -> klein_to -> logmap0
  float ko = 0.f;
  if (tid < DD) {
#pragma unroll
    for (int a = 0; a < AA; ++a) ko += kle[a * DP + tid] * pe[a];
  }
  float kn2 = block_reduce_sum((tid < DD) ? ko * ko : 0.f, red, tid);
  float f   = 1.f / (1.f + sqrtf(fmaxf(1.f - kn2, 0.f)));
  float rn  = sqrtf(kn2) * f;
  float rnc = fmaxf(rn, 1e-15f);
  float rc  = fminf(fmaxf(rn, -1.f + 1e-7f), 1.f - 1e-7f);
  float at  = 0.5f * logf((1.f + rc) / (1.f - rc));
  float coef = at / rnc * f;
  if (tid < DD) out_eu[(size_t)b * DD + tid] = coef * ko;
}

extern "C" void kernel_launch(void* const* d_in, const int* in_sizes, int n_in,
                              void* d_out, int out_size, void* d_ws, size_t ws_size,
                              hipStream_t stream) {
  (void)in_sizes; (void)n_in; (void)out_size; (void)ws_size;
  const int*   inp    = (const int*)d_in[0];
  const float* emb    = (const float*)d_in[1];
  const float* Mw     = (const float*)d_in[2];
  const float* a_emb  = (const float*)d_in[3];
  const float* seed_w = (const float*)d_in[4];
  float* out = (float*)d_out;

  float* ws   = (float*)d_ws;
  float* xavg = ws;                               // BB*DP
  float* Vm   = xavg + (size_t)BB * DP;           // BB*DP
  float* hyp  = Vm + (size_t)BB * DP;             // AA*DP
  float* kle  = hyp + AA * DP;                    // AA*DP
  float* q2v  = kle + AA * DP;                    // AA
  float* lfv  = q2v + AA;                         // AA
  float* Mp   = lfv + AA + 4;                     // DP*DP zero-padded M

  float* out_eu = out;
  float* out_pe = out + (size_t)BB * DD;
  float* out_ap = out_pe + (size_t)BB * AA;

  hipLaunchKernelGGL(k_anchors, dim3(AA), dim3(256), 0, stream, a_emb, seed_w, hyp, kle, q2v, lfv);
  hipLaunchKernelGGL(k_xavg, dim3(BB), dim3(256), 0, stream, inp, emb, xavg);
  hipLaunchKernelGGL(k_padM, dim3((DP * DP + 255) / 256), dim3(256), 0, stream, Mw, Mp);
  const int tiles = (BB / 16) * (DP / 16);
  hipLaunchKernelGGL(k_vgemm, dim3((tiles + 7) / 8), dim3(256), 0, stream, xavg, Mp, Vm);
  hipLaunchKernelGGL(k_final, dim3(BB), dim3(256), 0, stream, inp, emb, Vm, hyp, kle, q2v, lfv,
                     out_eu, out_pe, out_ap);
}
